// myGRUCell_diag_52235392254517
// MI455X (gfx1250) — matrix operand order repaired and checked by reading, not compiled, not run on hardware
//
#include <hip/hip_runtime.h>
#include <hip/hip_bf16.h>

// ---------------------------------------------------------------------------
// Low-rank GRU cell, fully fused for MI455X (gfx1250, wave32, WMMA bf16).
// One prep kernel (weight transpose->bf16), one fused main kernel.
// ---------------------------------------------------------------------------

typedef __attribute__((ext_vector_type(16))) __bf16 v16bf;
typedef __attribute__((ext_vector_type(8)))  __bf16 v8bf;
typedef __attribute__((ext_vector_type(8)))  float  v8f;
typedef __attribute__((ext_vector_type(4)))  float  f32x4;

#define BATCH 32768
#define HID   1024
#define RANK  64
#define LDP   1032   // padded bf16 row stride for 16x1024 LDS tiles (conflict-free b128 reads)
#define LDR   72     // padded bf16 row stride for 16x64 LDS tiles
#define LDF   68     // padded f32 row stride for the partial-sum tile

#define WMMA_BF16(A, B, C) \
  __builtin_amdgcn_wmma_f32_16x16x32_bf16(false, (A), false, (B), (short)0, (C), false, false)

static __device__ __forceinline__ float fast_sigmoid(float x) {
  return 1.f / (1.f + __expf(-x));
}
static __device__ __forceinline__ float fast_tanh(float x) {
  float t = __expf(-2.f * fabsf(x));
  float r = (1.f - t) / (1.f + t);
  return copysignf(r, x);
}

// A-fragment from bf16 row-major (LDS or global). Lane L: row = m, K runs
// k0+half*16..+15 (contiguous, matching load_b).
static __device__ __forceinline__ v16bf load_a_bf16(const __bf16* base, int ld, int m,
                                                    int k0, int half) {
  const __bf16* p = base + (size_t)m * ld + k0 + half * 16;
  v8bf lo = *(const v8bf*)(p);
  v8bf hi = *(const v8bf*)(p + 8);
  v16bf A;
#pragma unroll
  for (int i = 0; i < 8; ++i) { A[i] = lo[i]; A[8 + i] = hi[i]; }
  return A;
}

// B-fragment: weights pre-transposed as [N][K] bf16 row-major.
// Lane L: column N = nrow, K = k0 + half*16 .. +15 (contiguous 32B load).
static __device__ __forceinline__ v16bf load_b(const __bf16* baseT, int ld, int nrow,
                                               int k0, int half) {
  return *(const v16bf*)(baseT + (size_t)nrow * ld + k0 + half * 16);
}

// ---------------------------------------------------------------------------
// Kernel 0: convert + transpose all 8 weight matrices to bf16 [N][K] layout.
//   mats 0,1 : W,U      [1024][64]  -> [64][1024]
//   mats 2-7 : W1..U3   [64][1024]  -> [1024][64]
// ---------------------------------------------------------------------------
__global__ void prep_weights(const float* __restrict__ W,  const float* __restrict__ U,
                             const float* __restrict__ W1, const float* __restrict__ W2,
                             const float* __restrict__ W3, const float* __restrict__ U1,
                             const float* __restrict__ U2, const float* __restrict__ U3,
                             __bf16* __restrict__ ws) {
  int mat = blockIdx.y;
  int idx = blockIdx.x * blockDim.x + threadIdx.x;  // 0..65535
  const float* src;
  switch (mat) {
    case 0: src = W;  break; case 1: src = U;  break;
    case 2: src = W1; break; case 3: src = W2; break;
    case 4: src = W3; break; case 5: src = U1; break;
    case 6: src = U2; break; default: src = U3; break;
  }
  __bf16* dst = ws + (size_t)mat * 65536;
  int n, k, dld;
  if (mat < 2) { k = idx >> 6;  n = idx & 63;   dld = HID; }
  else         { k = idx >> 10; n = idx & 1023; dld = RANK; }
  dst[(size_t)n * dld + k] = (__bf16)src[idx];
}

// ---------------------------------------------------------------------------
// Fused kernel: per 16-row tile, the entire GRU cell.
//   Stage 0: x,h tiles -> LDS (bf16, coalesced, each HBM byte read once)
//   Phase 0: P = x@W, Q = h@U   (8 waves, one 16x16 tile each, K=1024 from LDS)
//   Phase 1: r,z gates; rh -> LDS (x buffer), cpre -> LDS (h buffer), z -> LDS
//   Phase 2: S = rh@U, K=1024 split across all 8 waves, partials reduced in LDS
//   Phase 3: c = tanh(cpre + S@U3); h_next = z*h + (1-z)*c
// All intermediate tiles live in LDS; per-lane register state stays < 256.
// ---------------------------------------------------------------------------
__global__ __launch_bounds__(256) void gru_fused(
    const float* __restrict__ x, const float* __restrict__ h,
    const __bf16* __restrict__ Wt,  const __bf16* __restrict__ Ut,
    const __bf16* __restrict__ W1t, const __bf16* __restrict__ W2t, const __bf16* __restrict__ W3t,
    const __bf16* __restrict__ U1t, const __bf16* __restrict__ U2t, const __bf16* __restrict__ U3t,
    const float* __restrict__ U1d, const float* __restrict__ U2d, const float* __restrict__ U3d,
    const float* __restrict__ br, const float* __restrict__ bg, const float* __restrict__ bu,
    float* __restrict__ out) {
  __shared__ __bf16 lds_x[16 * LDP];   // x tile; recycled as rh tile in phase 1
  __shared__ __bf16 lds_h[16 * LDP];   // h tile; recycled as cpre tile in phase 1
  __shared__ __bf16 lds_z[16 * LDP];   // update gate z tile
  __shared__ __bf16 lds_p[16 * LDR];   // P tile
  __shared__ __bf16 lds_q[16 * LDR];   // Q tile
  __shared__ __bf16 lds_s[16 * LDR];   // S tile
  __shared__ float  lds_part[16 * LDF];// f32 partial sums for phase-2 split-K

  const int tid  = threadIdx.x;
  const int wave = tid >> 5;
  const int lane = tid & 31;
  const int half = lane >> 4;
  const int lane16 = lane & 15;
  const int m0 = blockIdx.x * 16;

  // ---- Stage 0: cooperative coalesced stage+convert of x,h tiles into LDS ----
#pragma unroll
  for (int j = 0; j < 8; ++j) {
    const int e0  = (j * 256 + tid) * 8;     // 8 consecutive f32 per thread
    const int row = e0 >> 10;
    const int col = e0 & 1023;
    const size_t g = (size_t)(m0 + row) * HID + col;
    f32x4 a0 = *(const f32x4*)(x + g);
    f32x4 a1 = *(const f32x4*)(x + g + 4);
    f32x4 b0 = *(const f32x4*)(h + g);
    f32x4 b1 = *(const f32x4*)(h + g + 4);
    v8bf xa, hb;
#pragma unroll
    for (int i = 0; i < 4; ++i) {
      xa[i] = (__bf16)a0[i]; xa[4 + i] = (__bf16)a1[i];
      hb[i] = (__bf16)b0[i]; hb[4 + i] = (__bf16)b1[i];
    }
    *(v8bf*)(lds_x + row * LDP + col) = xa;
    *(v8bf*)(lds_h + row * LDP + col) = hb;
  }
  __syncthreads();

  // ---- Phase 0: P = x@W (waves 0-3), Q = h@U (waves 4-7); one 16x16 tile/wave ----
  {
    const __bf16* srcA = (wave < 4) ? lds_x : lds_h;
    const __bf16* srcB = (wave < 4) ? Wt : Ut;
    __bf16*       dst  = (wave < 4) ? lds_p : lds_q;
    const int n0 = (wave & 3) * 16;
    v8f acc = {};
#pragma unroll 4
    for (int k0 = 0; k0 < HID; k0 += 32) {
      v16bf a = load_a_bf16(srcA, LDP, lane16, k0, half);
      v16bf b = load_b(srcB, HID, n0 + lane16, k0, half);
      acc = WMMA_BF16(a, b, acc);
    }
#pragma unroll
    for (int v = 0; v < 8; ++v)
      dst[(v + half * 8) * LDR + n0 + lane16] = (__bf16)acc[v];
  }
  __syncthreads();

  // A-fragments over the K=64 rank dimension (reused across this wave's N-tiles).
  v16bf aP[2], aQ[2];
#pragma unroll
  for (int ks = 0; ks < 2; ++ks) {
    aP[ks] = load_a_bf16(lds_p, LDR, lane16, ks * 32, half);
    aQ[ks] = load_a_bf16(lds_q, LDR, lane16, ks * 32, half);
  }

  // ---- Phase 1: r, z gates; write rh / cpre / z tiles to LDS ----
  // (lds_h is dead after phase 0 -> recycled for cpre; lds_x -> rh.)
#pragma unroll
  for (int i = 0; i < 8; ++i) {
    const int n = (wave * 8 + i) * 16;
    const int col = n + lane16;
    v8f w1a = {}, w2a = {}, w3a = {}, u1a = {}, u2a = {};
#pragma unroll
    for (int ks = 0; ks < 2; ++ks) {
      const int k0 = ks * 32;
      w1a = WMMA_BF16(aP[ks], load_b(W1t, RANK, col, k0, half), w1a);
      w2a = WMMA_BF16(aP[ks], load_b(W2t, RANK, col, k0, half), w2a);
      w3a = WMMA_BF16(aP[ks], load_b(W3t, RANK, col, k0, half), w3a);
      u1a = WMMA_BF16(aQ[ks], load_b(U1t, RANK, col, k0, half), u1a);
      u2a = WMMA_BF16(aQ[ks], load_b(U2t, RANK, col, k0, half), u2a);
    }
    const float d1 = U1d[col], d2 = U2d[col], d3 = U3d[col];
    const float b1 = br[col],  b2 = bg[col],  b3 = bu[col];
#pragma unroll
    for (int v = 0; v < 8; ++v) {
      const int mrow = v + half * 8;
      const float hv = h[(size_t)(m0 + mrow) * HID + col];
      const float r  = fast_sigmoid(w1a[v] + u1a[v] + hv * d1 + b1);
      const float zv = fast_sigmoid(w2a[v] + u2a[v] + hv * d2 + b2);
      const float rh = r * hv;
      lds_x[mrow * LDP + col] = (__bf16)rh;                      // rh tile
      lds_h[mrow * LDP + col] = (__bf16)(w3a[v] + rh * d3 + b3); // cpre tile
      lds_z[mrow * LDP + col] = (__bf16)zv;                      // z tile
    }
  }
  __syncthreads();

  // ---- Phase 2: S = rh @ U, split-K across all 8 waves ----
  {
    const int nt = wave & 3;   // n-tile 0..3
    const int kh = wave >> 2;  // k-half 0..1
    v8f s = {};
#pragma unroll 4
    for (int ks = 0; ks < 16; ++ks) {
      const int k0 = kh * 512 + ks * 32;
      v16bf a = load_a_bf16(lds_x, LDP, lane16, k0, half);       // rh
      v16bf b = load_b(Ut, HID, nt * 16 + lane16, k0, half);
      s = WMMA_BF16(a, b, s);
    }
    if (kh == 1) {
#pragma unroll
      for (int v = 0; v < 8; ++v)
        lds_part[(v + half * 8) * LDF + nt * 16 + lane16] = s[v];
    }
    __syncthreads();
    if (kh == 0) {
#pragma unroll
      for (int v = 0; v < 8; ++v) {
        const float sv = s[v] + lds_part[(v + half * 8) * LDF + nt * 16 + lane16];
        lds_s[(v + half * 8) * LDR + nt * 16 + lane16] = (__bf16)sv;
      }
    }
  }
  __syncthreads();

  // ---- Phase 3: c = tanh(cpre + S @ U3); h_next = z*h + (1-z)*c ----
  v16bf aS[2];
#pragma unroll
  for (int ks = 0; ks < 2; ++ks)
    aS[ks] = load_a_bf16(lds_s, LDR, lane16, ks * 32, half);

#pragma unroll
  for (int i = 0; i < 8; ++i) {
    const int n = (wave * 8 + i) * 16;
    const int col = n + lane16;
    v8f acc = {};
#pragma unroll
    for (int ks = 0; ks < 2; ++ks)
      acc = WMMA_BF16(aS[ks], load_b(U3t, RANK, col, ks * 32, half), acc);
#pragma unroll
    for (int v = 0; v < 8; ++v) {
      const int mrow = v + half * 8;
      const float hv   = h[(size_t)(m0 + mrow) * HID + col];
      const float cp   = (float)lds_h[mrow * LDP + col];   // cpre
      const float zv   = (float)lds_z[mrow * LDP + col];   // z
      const float c    = fast_tanh(cp + acc[v]);
      out[(size_t)(m0 + mrow) * HID + col] = zv * hv + (1.f - zv) * c;
    }
  }
}

// ---------------------------------------------------------------------------
// Host launcher
// ---------------------------------------------------------------------------
extern "C" void kernel_launch(void* const* d_in, const int* in_sizes, int n_in,
                              void* d_out, int out_size, void* d_ws, size_t ws_size,
                              hipStream_t stream) {
  const float* x   = (const float*)d_in[0];
  const float* h   = (const float*)d_in[1];
  const float* W   = (const float*)d_in[2];
  const float* W1  = (const float*)d_in[3];
  const float* W2  = (const float*)d_in[4];
  const float* W3  = (const float*)d_in[5];
  const float* U   = (const float*)d_in[6];
  const float* U1  = (const float*)d_in[7];
  const float* U2  = (const float*)d_in[8];
  const float* U3  = (const float*)d_in[9];
  const float* U1d = (const float*)d_in[10];
  const float* U2d = (const float*)d_in[11];
  const float* U3d = (const float*)d_in[12];
  const float* br  = (const float*)d_in[13];
  const float* bg  = (const float*)d_in[14];
  const float* bu  = (const float*)d_in[15];
  float* out = (float*)d_out;

  // Workspace: 8 x 65536 bf16 transposed weights (1 MB total).
  __bf16* wt = (__bf16*)d_ws;
  const __bf16* Wt  = wt + 0 * 65536;
  const __bf16* Ut  = wt + 1 * 65536;
  const __bf16* W1t = wt + 2 * 65536;
  const __bf16* W2t = wt + 3 * 65536;
  const __bf16* W3t = wt + 4 * 65536;
  const __bf16* U1t = wt + 5 * 65536;
  const __bf16* U2t = wt + 6 * 65536;
  const __bf16* U3t = wt + 7 * 65536;

  prep_weights<<<dim3(256, 8), 256, 0, stream>>>(W, U, W1, W2, W3, U1, U2, U3, wt);
  gru_fused<<<BATCH / 16, 256, 0, stream>>>(x, h, Wt, Ut,
                                            W1t, W2t, W3t, U1t, U2t, U3t,
                                            U1d, U2d, U3d, br, bg, bu, out);
}
